// EdgePredictor_per_node_10788957848109
// MI455X (gfx1250) — compile-verified, matrix operand order checked by reading, not run on hardware
//
#include <hip/hip_runtime.h>
#include <hip/hip_bf16.h>

// ---------------- CDNA5 WMMA types ----------------
typedef __attribute__((ext_vector_type(16))) __bf16 v16bf;
typedef __attribute__((ext_vector_type(8)))  float  v8f;

#define D_DIM 256
#define HID   100
#define HPAD  112          // 7 WMMA N-tiles of 16
#define ROW_US 280         // bf16 elements per staged row (560B: 16B aligned, conflict-free stride)
#define SEG_US (16 * ROW_US)

__device__ __forceinline__ unsigned short f2bf(float f) {
    unsigned int u = __builtin_bit_cast(unsigned int, f);
    unsigned int r = u + 0x7FFFu + ((u >> 16) & 1u);   // round-to-nearest-even
    return (unsigned short)(r >> 16);
}

// ---------------- prep kernel 1: invert permutation ----------------
__global__ void prep_invperm(const int* __restrict__ src2, int* __restrict__ inv2, int E) {
    int j = blockIdx.x * blockDim.x + threadIdx.x;
    if (j < E) inv2[src2[j]] = j;
}

// ---------------- prep kernel 2: idx2, int64 idx outputs, W1->bf16 pad ----------------
__global__ void prep_rest(const int* __restrict__ src1, const int* __restrict__ inv2,
                          int* __restrict__ i2buf,
                          const float* __restrict__ W1, const float* __restrict__ b1,
                          const float* __restrict__ W2,
                          unsigned short* __restrict__ Wb,
                          float* __restrict__ b1p, float* __restrict__ w2p,
                          long long* __restrict__ idx1o, long long* __restrict__ idx2o,
                          int E) {
    int tid = blockIdx.x * blockDim.x + threadIdx.x;
    if (tid < HPAD * (3 * D_DIM)) {
        int n = tid / (3 * D_DIM);
        int k = tid - n * (3 * D_DIM);
        Wb[tid] = (n < HID) ? f2bf(W1[n * (3 * D_DIM) + k]) : (unsigned short)0;
    }
    if (tid < HPAD) {
        b1p[tid] = (tid < HID) ? b1[tid] : 0.0f;
        w2p[tid] = (tid < HID) ? W2[tid] : 0.0f;
    }
    if (tid < E) {
        int v = inv2[src1[tid]];
        i2buf[tid] = v;
        idx1o[tid] = (long long)tid;
        idx2o[tid] = (long long)v;
    }
}

// ---------------- LDS fragment loads ----------------
__device__ __forceinline__ v16bf load_a_frag(const unsigned short* sm, int seg, int kk,
                                             int er, int half) {
    // A 16x32 bf16 layout: lane(er,half) holds row er, K = kk*32 + half*8 + {0..7, 16..23}
    int base = seg * SEG_US + er * ROW_US + kk * 32 + half * 8;
    union { uint4 q[2]; v16bf v; } u;
    u.q[0] = *(const uint4*)(sm + base);
    u.q[1] = *(const uint4*)(sm + base + 16);
    return u.v;
}

__device__ __forceinline__ v16bf load_b_frag(const unsigned short* __restrict__ Wb,
                                             int ncol, int kbase) {
    // B 32x16 bf16 layout: lane = column ncol, K = kbase + {0..7, 16..23}
    const unsigned short* p = Wb + ncol * (3 * D_DIM) + kbase;
    union { uint4 q[2]; v16bf v; } u;
    u.q[0] = *(const uint4*)(p);
    u.q[1] = *(const uint4*)(p + 16);
    return u.v;
}

__device__ __forceinline__ void stage_row4(unsigned short* sm, int seg, int er, int c,
                                           float x, float y, float z, float w) {
    unsigned short* p = sm + seg * SEG_US + er * ROW_US + c;
    uint2 d;
    d.x = (unsigned)f2bf(x) | ((unsigned)f2bf(y) << 16);
    d.y = (unsigned)f2bf(z) | ((unsigned)f2bf(w) << 16);
    *(uint2*)p = d;
}

// ---------------- main kernel: 16 edges per wave32 block ----------------
__global__ void __launch_bounds__(32)
edge_mlp_wmma(const float* __restrict__ h, const int* __restrict__ i2buf,
              const unsigned short* __restrict__ Wb,
              const float* __restrict__ b1p, const float* __restrict__ w2p,
              const float* __restrict__ b2, float* __restrict__ outF, int E) {
    __shared__ __align__(16) unsigned short sm[5 * SEG_US];   // 44800 bytes

    const int lane = threadIdx.x;
    const int er   = lane & 15;
    const int half = lane >> 4;
    const int e0   = blockIdx.x * 16;

    // ---- stage activations (bf16) into LDS: each lane does half a row of all 5 segs ----
    {
        int e = e0 + er;
        if (e >= E) e = E - 1;
        const int i2 = i2buf[e];
        const float* rs1 = h + (size_t)e * D_DIM;                       // h_src1[e]
        const float* rs2 = h + ((size_t)3 * E + i2) * D_DIM;            // h_src2[i2]
        const float* rp1 = h + ((size_t)1 * E + e ) * D_DIM;            // h_pos_dst1[e]
        const float* rn1 = h + ((size_t)2 * E + e ) * D_DIM;            // h_neg_dst1[e]
        const float* rp2 = h + ((size_t)4 * E + i2) * D_DIM;            // h_pos_dst2[i2]
        const float* rn2 = h + ((size_t)5 * E + i2) * D_DIM;            // h_neg_dst2[i2]
        const int c0 = half * 128;
        for (int c = c0; c < c0 + 128; c += 4) {
            float4 a = *(const float4*)(rs1 + c);
            float4 b = *(const float4*)(rs2 + c);
            stage_row4(sm, 0, er, c, 0.5f * (a.x + b.x), 0.5f * (a.y + b.y),
                                     0.5f * (a.z + b.z), 0.5f * (a.w + b.w));
            float4 v1 = *(const float4*)(rp1 + c);
            stage_row4(sm, 1, er, c, v1.x, v1.y, v1.z, v1.w);
            float4 v2 = *(const float4*)(rn1 + c);
            stage_row4(sm, 2, er, c, v2.x, v2.y, v2.z, v2.w);
            float4 v3 = *(const float4*)(rp2 + c);
            stage_row4(sm, 3, er, c, v3.x, v3.y, v3.z, v3.w);
            float4 v4 = *(const float4*)(rn2 + c);
            stage_row4(sm, 4, er, c, v4.x, v4.y, v4.z, v4.w);
        }
    }
    __syncthreads();

    float outP[8], outA[8], outB[8], outC[8];
    #pragma unroll
    for (int r = 0; r < 8; ++r) { outP[r] = 0.f; outA[r] = 0.f; outB[r] = 0.f; outC[r] = 0.f; }

    const int nlocal = er;
    // ---- 7 N-tiles over padded hidden dim ----
    #pragma unroll 1
    for (int nb = 0; nb < 7; ++nb) {
        const int ncol = nb * 16 + nlocal;                 // hidden unit for this lane
        v8f accS  = {0.f,0.f,0.f,0.f,0.f,0.f,0.f,0.f};
        v8f accP1 = {0.f,0.f,0.f,0.f,0.f,0.f,0.f,0.f};
        v8f accN1 = {0.f,0.f,0.f,0.f,0.f,0.f,0.f,0.f};
        v8f accP2 = {0.f,0.f,0.f,0.f,0.f,0.f,0.f,0.f};
        v8f accN2 = {0.f,0.f,0.f,0.f,0.f,0.f,0.f,0.f};
        #pragma unroll
        for (int kk = 0; kk < 8; ++kk) {
            const int kb = kk * 32 + half * 8;
            v16bf aS  = load_a_frag(sm, 0, kk, er, half);
            v16bf aP1 = load_a_frag(sm, 1, kk, er, half);
            v16bf aN1 = load_a_frag(sm, 2, kk, er, half);
            v16bf aP2 = load_a_frag(sm, 3, kk, er, half);
            v16bf aN2 = load_a_frag(sm, 4, kk, er, half);
            v16bf bS  = load_b_frag(Wb, ncol,       0 + kb);   // W1[:, 0:256]
            v16bf bM  = load_b_frag(Wb, ncol,  D_DIM + kb);    // W1[:, 256:512] (shared p1/n1)
            v16bf bL  = load_b_frag(Wb, ncol, 2*D_DIM + kb);   // W1[:, 512:768] (shared p2/n2)
            accS  = __builtin_amdgcn_wmma_f32_16x16x32_bf16(false, aS,  false, bS, (short)0, accS,  false, false);
            accP1 = __builtin_amdgcn_wmma_f32_16x16x32_bf16(false, aP1, false, bM, (short)0, accP1, false, false);
            accN1 = __builtin_amdgcn_wmma_f32_16x16x32_bf16(false, aN1, false, bM, (short)0, accN1, false, false);
            accP2 = __builtin_amdgcn_wmma_f32_16x16x32_bf16(false, aP2, false, bL, (short)0, accP2, false, false);
            accN2 = __builtin_amdgcn_wmma_f32_16x16x32_bf16(false, aN2, false, bL, (short)0, accN2, false, false);
        }
        const float bb = b1p[ncol];
        const float ww = w2p[ncol];
        #pragma unroll
        for (int r = 0; r < 8; ++r) {
            const float s  = accS[r];
            const float p1 = accP1[r], n1 = accN1[r];
            const float p2 = accP2[r], n2 = accN2[r];
            outP[r] += fmaxf(s + p1 + p2 + bb, 0.f) * ww;   // pred_pos
            outA[r] += fmaxf(s + n1 + p2 + bb, 0.f) * ww;   // neg block A
            outB[r] += fmaxf(s + p1 + n2 + bb, 0.f) * ww;   // neg block B
            outC[r] += fmaxf(s + n1 + n2 + bb, 0.f) * ww;   // neg block C
        }
    }

    // ---- reduce over the 16 hidden columns held across each 16-lane half ----
    const float b2v = b2[0];
    #pragma unroll
    for (int r = 0; r < 8; ++r) {
        float vP = outP[r], vA = outA[r], vB = outB[r], vC = outC[r];
        #pragma unroll
        for (int m = 8; m >= 1; m >>= 1) {
            vP += __shfl_xor(vP, m, 16);
            vA += __shfl_xor(vA, m, 16);
            vB += __shfl_xor(vB, m, 16);
            vC += __shfl_xor(vC, m, 16);
        }
        if ((lane & 15) == 0) {
            const int e = e0 + r + half * 8;               // C-layout: VGPR r -> row r (+8 upper half)
            if (e < E) {
                outF[e]            = vP + b2v;             // pred_true (all labels==1)
                outF[(size_t)E + e]      = vA + b2v;       // pred_neg block A
                outF[(size_t)2 * E + e]  = vB + b2v;       // pred_neg block B
                outF[(size_t)3 * E + e]  = vC + b2v;       // pred_neg block C
            }
        }
    }
}

// ---------------- host launcher ----------------
extern "C" void kernel_launch(void* const* d_in, const int* in_sizes, int n_in,
                              void* d_out, int out_size, void* d_ws, size_t ws_size,
                              hipStream_t stream) {
    const float* h    = (const float*)d_in[0];
    const int*   src1 = (const int*)d_in[3];
    const int*   src2 = (const int*)d_in[4];
    const float* W1   = (const float*)d_in[5];
    const float* b1   = (const float*)d_in[6];
    const float* W2   = (const float*)d_in[7];
    const float* b2   = (const float*)d_in[8];
    const int E = in_sizes[1];                 // NUM_EDGE (pos_edge_label length)

    // workspace carve-up
    char* ws = (char*)d_ws;
    int*            inv2 = (int*)ws;                              // E ints
    int*            i2b  = (int*)(ws + (size_t)E * 4);            // E ints
    unsigned short* Wb   = (unsigned short*)(ws + (size_t)2 * E * 4);          // 112*768 bf16
    float*          b1p  = (float*)(ws + (size_t)2 * E * 4 + HPAD * 3 * D_DIM * 2);
    float*          w2p  = b1p + HPAD;

    float*     outF  = (float*)d_out;
    long long* idx1o = (long long*)(outF + (size_t)4 * E);
    long long* idx2o = idx1o + E;

    const int TB = 256;
    prep_invperm<<<(E + TB - 1) / TB, TB, 0, stream>>>(src2, inv2, E);

    const int N2 = HPAD * 3 * D_DIM;           // 86016 > E
    const int n2grid = ((N2 > E ? N2 : E) + TB - 1) / TB;
    prep_rest<<<n2grid, TB, 0, stream>>>(src1, inv2, i2b, W1, b1, W2,
                                         Wb, b1p, w2p, idx1o, idx2o, E);

    const int nblocks = (E + 15) / 16;         // one wave32 per block, 16 edges each
    edge_mlp_wmma<<<nblocks, 32, 0, stream>>>(h, i2b, Wb, b1p, w2p, b2, outF, E);
}
